// RecurrentDecoder_14061722927624
// MI455X (gfx1250) — compile-verified, matrix operand order checked by reading, not run on hardware
//
#include <hip/hip_runtime.h>

#define BDIM 1024
#define HDIM 1024
#define FDIM 128
#define TLEN 256

typedef __attribute__((ext_vector_type(16))) __bf16 v16bf;
typedef __attribute__((ext_vector_type(8)))  float  v8f;

union TileU { v16bf v; uint4 u[2]; };

__device__ __forceinline__ unsigned short f2bf(float f) {
  unsigned int u = __builtin_bit_cast(unsigned int, f);
  u += 0x7fffu + ((u >> 16) & 1u);          // round-to-nearest-even
  return (unsigned short)(u >> 16);
}

__device__ __forceinline__ float fsigmoid(float x) {
  return 1.0f / (1.0f + __expf(-x));
}
__device__ __forceinline__ float ftanh_fast(float x) {
  return 1.0f - 2.0f / (__expf(2.0f * x) + 1.0f);
}

// Load a 16x32 bf16 tile (WMMA A/B operand layout) from a row-major
// [rows x pitch] array of bf16 (stored as ushort).
// Lanes 0-15:  row = lane,     halves hold K = k0+0..7  and k0+16..23
// Lanes 16-31: row = lane-16,  halves hold K = k0+8..15 and k0+24..31
__device__ __forceinline__ v16bf load_tile(const unsigned short* __restrict__ base,
                                           int row0, int pitch, int k0, int lane) {
  int r  = lane & 15;
  int kh = (lane & 16) ? 8 : 0;
  const unsigned short* p = base + (size_t)(row0 + r) * pitch + (k0 + kh);
  TileU t;
  t.u[0] = *(const uint4*)(p);
  t.u[1] = *(const uint4*)(p + 16);
  return t.v;
}

__device__ __forceinline__ v8f bf16_wmma(v16bf a, v16bf b, v8f c) {
  return __builtin_amdgcn_wmma_f32_16x16x32_bf16(
      /*neg_a=*/false, a, /*neg_b=*/false, b,
      /*c_mod=*/(short)0, c, /*reuse_a=*/false, /*reuse_b=*/false);
}

// ---------------------------------------------------------------------------
// f32 -> bf16 conversion (optionally mirroring the f32 value)
// ---------------------------------------------------------------------------
__global__ void convert_f32_bf16(const float* __restrict__ src,
                                 unsigned short* __restrict__ dst,
                                 float* __restrict__ dst_f32, int n) {
  int i = blockIdx.x * blockDim.x + threadIdx.x;
  if (i < n) {
    float v = src[i];
    dst[i] = f2bf(v);
    if (dst_f32) dst_f32[i] = v;
  }
}

// ---------------------------------------------------------------------------
// x = h @ W_dec^T + b_dec ; writes bf16 x for the next GRU step and the f32
// output slice out[b, t*F + f].  Each wave computes TWO 16x16 m-tiles that
// share one W_dec tile (2 WMMAs per 3 tile loads).
// Grid: 32 blocks x 256 threads (8 waves).
// ---------------------------------------------------------------------------
__global__ void __launch_bounds__(256)
dec_kernel(const unsigned short* __restrict__ hb,    // [B x H] bf16
           const unsigned short* __restrict__ wdec,  // [F x H] bf16 (N-major)
           const float* __restrict__ bdec,           // [F]
           unsigned short* __restrict__ xb,          // [B x F] bf16
           float* __restrict__ outp,                 // [B x T*F]
           int t) {
  int lane = threadIdx.x & 31;
  int wave = threadIdx.x >> 5;
  int tile = blockIdx.x * 8 + wave;   // 256 wave-jobs
  int nt    = tile & 7;               // F/16 = 8
  int mpair = tile >> 3;              // 32 m-pairs (B/32)
  int mt0 = mpair * 2;

  v8f acc0 = {}, acc1 = {};
#pragma unroll 2
  for (int k = 0; k < HDIM; k += 32) {
    v16bf a0 = load_tile(hb, mt0 * 16,      HDIM, k, lane);
    v16bf a1 = load_tile(hb, mt0 * 16 + 16, HDIM, k, lane);
    v16bf b  = load_tile(wdec, nt * 16,     HDIM, k, lane);
    acc0 = bf16_wmma(a0, b, acc0);
    acc1 = bf16_wmma(a1, b, acc1);
  }

  int n   = lane & 15;
  int mhi = (lane >> 4) << 3;
  int f   = nt * 16 + n;
  float bias = bdec[f];
#pragma unroll
  for (int i = 0; i < 2; ++i) {
    v8f acc = (i == 0) ? acc0 : acc1;
#pragma unroll
    for (int r = 0; r < 8; ++r) {
      int bm = (mt0 + i) * 16 + mhi + r;
      float v = acc[r] + bias;
      xb[(size_t)bm * FDIM + f] = f2bf(v);
      outp[(size_t)bm * (TLEN * FDIM) + t * FDIM + f] = v;
    }
  }
}

// ---------------------------------------------------------------------------
// One GRU step: h_new = (1-z)*n + z*h  with
//   r = sig(x W_ih_r^T + h W_hh_r^T + b), z likewise,
//   n = tanh((x W_ih_n^T + b_ih_n) + r*(h W_hh_n^T + b_hh_n))
// Each wave computes TWO 16x16 m-tiles of h_new sharing the 3 gate B tiles
// (6 WMMAs per 5 tile loads).  A block's 8 waves cover 16 consecutive
// m-tiles at one n-tile so B operands are shared through WGP$/L2.
// Grid: 256 blocks x 256 threads.
// ---------------------------------------------------------------------------
__global__ void __launch_bounds__(256)
gru_kernel(const unsigned short* __restrict__ xb,     // [B x F] bf16
           const float* __restrict__ hf_in,           // [B x H] f32
           const unsigned short* __restrict__ hb_in,  // [B x H] bf16
           const unsigned short* __restrict__ wih,    // [3H x F] bf16
           const unsigned short* __restrict__ whh,    // [3H x H] bf16
           const float* __restrict__ bih,             // [3H]
           const float* __restrict__ bhh,             // [3H]
           float* __restrict__ hf_out,                // [B x H] f32
           unsigned short* __restrict__ hb_out) {     // [B x H] bf16
  int lane = threadIdx.x & 31;
  int wave = threadIdx.x >> 5;
  int nt  = blockIdx.x & 63;                       // H/16 = 64 n-tiles
  int mt0 = ((blockIdx.x >> 6) * 16) + wave * 2;   // 64 m-tiles, 2 per wave

  v8f acc_r0 = {}, acc_z0 = {}, acc_in0 = {}, acc_hn0 = {};
  v8f acc_r1 = {}, acc_z1 = {}, acc_in1 = {}, acc_hn1 = {};

  // h @ W_hh^T contribution (K = 1024): 6 WMMAs per 32-wide K chunk
#pragma unroll 2
  for (int k = 0; k < HDIM; k += 32) {
    v16bf a0 = load_tile(hb_in, mt0 * 16,      HDIM, k, lane);
    v16bf a1 = load_tile(hb_in, mt0 * 16 + 16, HDIM, k, lane);
    v16bf br = load_tile(whh,            nt * 16, HDIM, k, lane);
    v16bf bz = load_tile(whh, HDIM     + nt * 16, HDIM, k, lane);
    v16bf bn = load_tile(whh, 2 * HDIM + nt * 16, HDIM, k, lane);
    acc_r0  = bf16_wmma(a0, br, acc_r0);
    acc_r1  = bf16_wmma(a1, br, acc_r1);
    acc_z0  = bf16_wmma(a0, bz, acc_z0);
    acc_z1  = bf16_wmma(a1, bz, acc_z1);
    acc_hn0 = bf16_wmma(a0, bn, acc_hn0);
    acc_hn1 = bf16_wmma(a1, bn, acc_hn1);
  }

  // x @ W_ih^T contribution (K = 128)
#pragma unroll 2
  for (int k = 0; k < FDIM; k += 32) {
    v16bf a0 = load_tile(xb, mt0 * 16,      FDIM, k, lane);
    v16bf a1 = load_tile(xb, mt0 * 16 + 16, FDIM, k, lane);
    v16bf br = load_tile(wih,            nt * 16, FDIM, k, lane);
    v16bf bz = load_tile(wih, HDIM     + nt * 16, FDIM, k, lane);
    v16bf bn = load_tile(wih, 2 * HDIM + nt * 16, FDIM, k, lane);
    acc_r0  = bf16_wmma(a0, br, acc_r0);
    acc_r1  = bf16_wmma(a1, br, acc_r1);
    acc_z0  = bf16_wmma(a0, bz, acc_z0);
    acc_z1  = bf16_wmma(a1, bz, acc_z1);
    acc_in0 = bf16_wmma(a0, bn, acc_in0);
    acc_in1 = bf16_wmma(a1, bn, acc_in1);
  }

  int n   = lane & 15;
  int mhi = (lane >> 4) << 3;
  int jn  = nt * 16 + n;
  float br_b  = bih[jn] + bhh[jn];
  float bz_b  = bih[HDIM + jn] + bhh[HDIM + jn];
  float bin_b = bih[2 * HDIM + jn];
  float bhn_b = bhh[2 * HDIM + jn];

#pragma unroll
  for (int i = 0; i < 2; ++i) {
    v8f ar = (i == 0) ? acc_r0  : acc_r1;
    v8f az = (i == 0) ? acc_z0  : acc_z1;
    v8f ai = (i == 0) ? acc_in0 : acc_in1;
    v8f ah = (i == 0) ? acc_hn0 : acc_hn1;
#pragma unroll
    for (int r8 = 0; r8 < 8; ++r8) {
      int bm = (mt0 + i) * 16 + mhi + r8;
      float rg = fsigmoid(ar[r8] + br_b);
      float zg = fsigmoid(az[r8] + bz_b);
      float ng = ftanh_fast(ai[r8] + bin_b + rg * (ah[r8] + bhn_b));
      float hprev = hf_in[(size_t)bm * HDIM + jn];
      float hnew  = (1.0f - zg) * ng + zg * hprev;
      hf_out[(size_t)bm * HDIM + jn] = hnew;
      hb_out[(size_t)bm * HDIM + jn] = f2bf(hnew);
    }
  }
}

// ---------------------------------------------------------------------------
extern "C" void kernel_launch(void* const* d_in, const int* in_sizes, int n_in,
                              void* d_out, int out_size, void* d_ws, size_t ws_size,
                              hipStream_t stream) {
  const float* h0   = (const float*)d_in[0];  // [1, B, H]
  const float* Wih  = (const float*)d_in[1];  // [3H, F]
  const float* Whh  = (const float*)d_in[2];  // [3H, H]
  const float* bih  = (const float*)d_in[3];  // [3H]
  const float* bhh  = (const float*)d_in[4];  // [3H]
  const float* Wdec = (const float*)d_in[5];  // [F, H]
  const float* bdec = (const float*)d_in[6];  // [F]
  float* out = (float*)d_out;                 // [B, T*F]
  (void)in_sizes; (void)n_in; (void)out_size; (void)ws_size;

  // Workspace carve-up (256B aligned), ~20.4 MB total
  char* ws = (char*)d_ws;
  size_t off = 0;
  auto carve = [&](size_t bytes) -> char* {
    char* p = ws + off;
    off += (bytes + 255) & ~(size_t)255;
    return p;
  };
  unsigned short* wih_b  = (unsigned short*)carve((size_t)3 * HDIM * FDIM * 2);
  unsigned short* whh_b  = (unsigned short*)carve((size_t)3 * HDIM * HDIM * 2);
  unsigned short* wdec_b = (unsigned short*)carve((size_t)FDIM * HDIM * 2);
  float*          hf[2];
  unsigned short* hb[2];
  unsigned short* xxb[2];
  for (int i = 0; i < 2; ++i) {
    hf[i]  = (float*)carve((size_t)BDIM * HDIM * 4);
    hb[i]  = (unsigned short*)carve((size_t)BDIM * HDIM * 2);
    xxb[i] = (unsigned short*)carve((size_t)BDIM * FDIM * 2);
  }

  // Weight + initial-state conversion to bf16
  {
    int n1 = 3 * HDIM * FDIM;
    convert_f32_bf16<<<(n1 + 255) / 256, 256, 0, stream>>>(Wih, wih_b, nullptr, n1);
    int n2 = 3 * HDIM * HDIM;
    convert_f32_bf16<<<(n2 + 255) / 256, 256, 0, stream>>>(Whh, whh_b, nullptr, n2);
    int n3 = FDIM * HDIM;
    convert_f32_bf16<<<(n3 + 255) / 256, 256, 0, stream>>>(Wdec, wdec_b, nullptr, n3);
    int n4 = BDIM * HDIM;
    convert_f32_bf16<<<(n4 + 255) / 256, 256, 0, stream>>>(h0, hb[0], hf[0], n4);
  }

  // Initial decode: x0 = h0 @ Wdec^T + b.  It writes the t=0 slice of `out`
  // as a scratch target; the real step-0 decode below overwrites it (stream
  // ordering makes this deterministic) -- keeps the epilogue branch-free.
  dec_kernel<<<32, 256, 0, stream>>>(hb[0], wdec_b, bdec, xxb[0], out, 0);

  // Sequential scan: h = GRU(x, h); x = h @ Wdec^T + b; emit x at step t
  for (int t = 0; t < TLEN; ++t) {
    int cur = t & 1, nxt = cur ^ 1;
    gru_kernel<<<256, 256, 0, stream>>>(xxb[cur], hf[cur], hb[cur],
                                        wih_b, whh_b, bih, bhh,
                                        hf[nxt], hb[nxt]);
    dec_kernel<<<32, 256, 0, stream>>>(hb[nxt], wdec_b, bdec, xxb[nxt], out, t);
  }
}
